// GumbelSlotSelector_4028679323694
// MI455X (gfx1250) — matrix operand order repaired and checked by reading, not compiled, not run on hardware
//
#include <hip/hip_runtime.h>
#include <hip/hip_bf16.h>

typedef __attribute__((ext_vector_type(16))) __bf16 v16bf;
typedef __attribute__((ext_vector_type(8)))  __bf16 v8bf;
typedef __attribute__((ext_vector_type(8)))  float  v8f;

#define D_DIM    256
#define H_DIM    256
#define K_SLOTS  512
#define B_BATCH  256
#define M_TOTAL  (B_BATCH * K_SLOTS)   /* 131072 rows */
#define MT       64                    /* rows per block */
#define LDSA_STR 264                   /* padded bf16 row stride (528B, 16B aligned) */

static __device__ __forceinline__ __bf16 f2bf(float f) {
    union { float f; unsigned u; } a; a.f = f;
    unsigned r = a.u + 0x7FFFu + ((a.u >> 16) & 1u);   // round-to-nearest-even
    union { unsigned short s; __bf16 b; } c;
    c.s = (unsigned short)(r >> 16);
    return c.b;
}

union FragU { v16bf v; v8bf h[2]; };

// ---------------------------------------------------------------------------
// Pack w1 (D x H, row-major f32) into per-lane WMMA B-fragment order (bf16).
// Layout: [kk(8)][nt(16)][lane(32)][16 halves], element e of lane ->
//   K = kk*32 + (lane<16 ? 0 : 16) + e,  N = nt*16 + (lane & 15).
// ---------------------------------------------------------------------------
__global__ __launch_bounds__(256) void pack_w1_kernel(const float* __restrict__ w1,
                                                      __bf16* __restrict__ Bp) {
    int t = blockIdx.x * 256 + threadIdx.x;       // 4096 threads total
    if (t >= 8 * 16 * 32) return;
    int lane = t & 31;
    int nt   = (t >> 5) & 15;
    int kk   = t >> 9;
    int n    = nt * 16 + (lane & 15);
    int kb   = kk * 32 + ((lane & 16) ? 16 : 0);
    __bf16 vals[16];
#pragma unroll
    for (int e = 0; e < 16; ++e)
        vals[e] = f2bf(w1[(size_t)(kb + e) * H_DIM + n]);
    __bf16* dst = Bp + (((size_t)(kk * 16 + nt) * 32 + lane) * 16);
    *(v8bf*)dst       = *(v8bf*)&vals[0];
    *(v8bf*)(dst + 8) = *(v8bf*)&vals[8];
}

// ---------------------------------------------------------------------------
// Fused: LayerNorm -> bf16 WMMA GEMM (D->H) + bias + ReLU -> (H->2) -> gumbel.
// Block: 256 threads (8 waves), 64 rows. Wave w: LN rows w*8..w*8+7, then
// GEMM tile rows (w&3)*16..+15, cols (w>>2)*128..+127.
// ---------------------------------------------------------------------------
__global__ __launch_bounds__(256) void fused_kernel(
    const float* __restrict__ slots, const float* __restrict__ gamma,
    const float* __restrict__ beta,  const __bf16* __restrict__ Bp,
    const float* __restrict__ b1,    const float* __restrict__ w2,
    const float* __restrict__ b2,    const float* __restrict__ u,
    float* __restrict__ out)
{
    __shared__ __align__(16) __bf16 ldsA[MT][LDSA_STR];
    __shared__ float ldsLog[MT][2][2];

    const int tid  = threadIdx.x;
    const int w    = tid >> 5;
    const int lane = tid & 31;
    const int mb   = blockIdx.x;

    // ---------------- LayerNorm (each wave: 8 rows, 8 cols/lane) -----------
    const int c0 = lane * 8;
    float ga[8], be[8];
    {
        float4 g0 = *(const float4*)(gamma + c0);
        float4 g1 = *(const float4*)(gamma + c0 + 4);
        float4 t0 = *(const float4*)(beta  + c0);
        float4 t1 = *(const float4*)(beta  + c0 + 4);
        ga[0]=g0.x; ga[1]=g0.y; ga[2]=g0.z; ga[3]=g0.w;
        ga[4]=g1.x; ga[5]=g1.y; ga[6]=g1.z; ga[7]=g1.w;
        be[0]=t0.x; be[1]=t0.y; be[2]=t0.z; be[3]=t0.w;
        be[4]=t1.x; be[5]=t1.y; be[6]=t1.z; be[7]=t1.w;
    }
#pragma unroll
    for (int i = 0; i < 8; ++i) {
        const int rloc = w * 8 + i;
        const float* row = slots + (size_t)(mb * MT + rloc) * D_DIM;
        float4 a0 = *(const float4*)(row + c0);
        float4 a1 = *(const float4*)(row + c0 + 4);
        float x[8] = {a0.x, a0.y, a0.z, a0.w, a1.x, a1.y, a1.z, a1.w};
        float s = 0.f, ss = 0.f;
#pragma unroll
        for (int j = 0; j < 8; ++j) { s += x[j]; ss += x[j] * x[j]; }
#pragma unroll
        for (int m = 1; m < 32; m <<= 1) {
            s  += __shfl_xor(s,  m, 32);
            ss += __shfl_xor(ss, m, 32);
        }
        float mean = s  * (1.0f / 256.0f);
        float var  = ss * (1.0f / 256.0f) - mean * mean;
        float rstd = rsqrtf(var + 1e-5f);
        __bf16 v[8];
#pragma unroll
        for (int j = 0; j < 8; ++j)
            v[j] = f2bf(((x[j] - mean) * rstd) * ga[j] + be[j]);
        *(v8bf*)&ldsA[rloc][c0] = *(v8bf*)&v[0];
    }
    __syncthreads();

    // ---------------- WMMA GEMM: 16 rows x 128 cols per wave ---------------
    const int mg   = w & 3;
    const int nh   = w >> 2;
    const int mrow = mg * 16 + (lane & 15);
    v8f acc[8] = {};
    for (int kk = 0; kk < 8; ++kk) {
        // A fragment (contiguous, matches packed B): lanes<16 -> K {0..15}, lanes>=16 -> K {16..31}
        const __bf16* pa = &ldsA[mrow][kk * 32 + ((lane & 16) ? 16 : 0)];
        FragU af;
        af.h[0] = *(const v8bf*)pa;
        af.h[1] = *(const v8bf*)(pa + 8);
#pragma unroll
        for (int nt = 0; nt < 8; ++nt) {
            const __bf16* pb =
                Bp + (((size_t)(kk * 16 + (nh * 8 + nt)) * 32 + lane) * 16);
            FragU bf;
            bf.h[0] = *(const v8bf*)pb;
            bf.h[1] = *(const v8bf*)(pb + 8);
            acc[nt] = __builtin_amdgcn_wmma_f32_16x16x32_bf16(
                false, af.v, false, bf.v, (short)0, acc[nt], false, false);
        }
    }

    // ---------------- epilogue: bias + ReLU + (H -> 2) ---------------------
    float l0r[8] = {0, 0, 0, 0, 0, 0, 0, 0};
    float l1r[8] = {0, 0, 0, 0, 0, 0, 0, 0};
#pragma unroll
    for (int nt = 0; nt < 8; ++nt) {
        int n_g   = (nh * 8 + nt) * 16 + (lane & 15);
        float bb  = b1[n_g];
        float w20 = w2[n_g * 2 + 0];
        float w21 = w2[n_g * 2 + 1];
#pragma unroll
        for (int r = 0; r < 8; ++r) {
            float hv = fmaxf(acc[nt][r] + bb, 0.0f);
            l0r[r] += hv * w20;
            l1r[r] += hv * w21;
        }
    }
#pragma unroll
    for (int r = 0; r < 8; ++r) {
#pragma unroll
        for (int m = 1; m < 16; m <<= 1) {
            l0r[r] += __shfl_xor(l0r[r], m, 32);
            l1r[r] += __shfl_xor(l1r[r], m, 32);
        }
    }
    if (lane == 0 || lane == 16) {
        int rbase = mg * 16 + ((lane == 16) ? 8 : 0);
#pragma unroll
        for (int r = 0; r < 8; ++r) {
            ldsLog[rbase + r][nh][0] = l0r[r];
            ldsLog[rbase + r][nh][1] = l1r[r];
        }
    }
    __syncthreads();

    // ---------------- gumbel finalize: one thread per row ------------------
    if (tid < MT) {
        int gm   = mb * MT + tid;
        float l0 = ldsLog[tid][0][0] + ldsLog[tid][1][0] + b2[0];
        float l1 = ldsLog[tid][0][1] + ldsLog[tid][1][1] + b2[1];
        float u0 = u[(size_t)gm * 2 + 0];
        float u1 = u[(size_t)gm * 2 + 1];
        float g0 = -logf(-logf(u0));
        float g1 = -logf(-logf(u1));
        float hard = ((l1 + g1) > (l0 + g0)) ? 1.0f : 0.0f;  // argmax==1 iff strictly greater
        float soft = 1.0f / (1.0f + expf(l0 - l1));           // softmax(...)[1]
        out[gm]           = hard;
        out[M_TOTAL + gm] = soft;
    }
}

// ---------------------------------------------------------------------------
// Lower-bound resampling: if a batch row kept 0 slots, promote the dropped
// slot with the largest rand_key (strict '>' keeps the lowest index on ties,
// matching argsort stability).
// ---------------------------------------------------------------------------
__global__ __launch_bounds__(512) void resample_kernel(float* __restrict__ out,
                                                       const float* __restrict__ rk) {
    __shared__ float sval[K_SLOTS];
    __shared__ int   sidx[K_SLOTS];
    __shared__ int   skept;
    const int b = blockIdx.x;
    const int k = threadIdx.x;
    if (k == 0) skept = 0;
    __syncthreads();
    float hard = out[(size_t)b * K_SLOTS + k];
    if (hard != 0.0f) atomicAdd(&skept, 1);
    sval[k] = rk[(size_t)b * K_SLOTS + k];
    sidx[k] = k;
    __syncthreads();
    for (int s = K_SLOTS / 2; s > 0; s >>= 1) {
        if (k < s) {
            if (sval[k + s] > sval[k]) { sval[k] = sval[k + s]; sidx[k] = sidx[k + s]; }
        }
        __syncthreads();
    }
    if (k == 0 && skept == 0)
        out[(size_t)b * K_SLOTS + sidx[0]] = 1.0f;
}

extern "C" void kernel_launch(void* const* d_in, const int* in_sizes, int n_in,
                              void* d_out, int out_size, void* d_ws, size_t ws_size,
                              hipStream_t stream) {
    const float* slots = (const float*)d_in[0];
    const float* gamma = (const float*)d_in[1];
    const float* beta  = (const float*)d_in[2];
    const float* w1    = (const float*)d_in[3];
    const float* b1    = (const float*)d_in[4];
    const float* w2    = (const float*)d_in[5];
    const float* b2    = (const float*)d_in[6];
    const float* u     = (const float*)d_in[7];
    const float* rk    = (const float*)d_in[8];
    float*  out = (float*)d_out;
    __bf16* Bp  = (__bf16*)d_ws;               // 128 KB packed w1 fragments

    pack_w1_kernel<<<16, 256, 0, stream>>>(w1, Bp);
    fused_kernel<<<M_TOTAL / MT, 256, 0, stream>>>(slots, gamma, beta, Bp,
                                                   b1, w2, b2, u, out);
    resample_kernel<<<B_BATCH, 512, 0, stream>>>(out, rk);
}
